// head_8985071583707
// MI455X (gfx1250) — matrix operand order repaired and checked by reading, not compiled, not run on hardware
//
#include <hip/hip_runtime.h>
#include <hip/hip_bf16.h>

// Problem sizes (fixed by the reference)
#define NB   128      // batch
#define TT   512      // sequence length
#define CC   384      // n_embed
#define DD   64       // head size

typedef __attribute__((ext_vector_type(16))) __bf16         v16bf;
typedef __attribute__((ext_vector_type(16))) unsigned short v16us;
typedef __attribute__((ext_vector_type(8)))  float          v8f;
typedef __attribute__((ext_vector_type(4)))  unsigned int   v4u;
typedef __attribute__((ext_vector_type(8)))  int            v8i;
typedef __attribute__((ext_vector_type(4)))  int            v4i;

#if __has_builtin(__builtin_amdgcn_tensor_load_to_lds)
#define USE_TDM 1
#else
#define USE_TDM 0
#endif

union Frag {
  v16us u;
  v16bf b;
  uint4 x[2];
};

__device__ __forceinline__ unsigned short f2bf(float f) {
  unsigned u = __builtin_bit_cast(unsigned, f);
  return (unsigned short)((u + 0x7FFFu + ((u >> 16) & 1u)) >> 16);  // RNE
}

__device__ __forceinline__ v8f wmma_bf16(v16bf a, v16bf b, v8f c) {
  // D = A(16x32 bf16) * B(32x16 bf16) + C(16x16 f32)
  return __builtin_amdgcn_wmma_f32_16x16x32_bf16(false, a, false, b, (short)0, c,
                                                 false, false);
}

// A-operand: 32 contiguous bf16 K-values at `base`.
// element e -> K = (e>>3)*16 + h*8 + (e&7)   (h = lane>>4)
__device__ __forceinline__ Frag load_a_bf16(const unsigned short* base, int h) {
  const uint4* p = reinterpret_cast<const uint4*>(base);
  Frag f; f.x[0] = p[h]; f.x[1] = p[2 + h]; return f;
}
// B-operand: 32 contiguous bf16 K-values at `base`.
// element e -> K = (e>>3)*16 + h*8 + (e&7)   (h = lane>>4)
__device__ __forceinline__ Frag load_b_bf16(const unsigned short* base, int h) {
  const uint4* p = reinterpret_cast<const uint4*>(base);
  Frag f; f.x[0] = p[h]; f.x[1] = p[2 + h]; return f;
}

#if USE_TDM
// TDM: DMA a 2D bf16 tile (tile_dim0=32 cols, tile_dim1=192 rows, row stride 384
// elements) from global memory into LDS at lds_off. Descriptor per CDNA5 ISA
// "08_async_tensor" sec 8.3/8.4 (D# group0 128b, group1 256b; groups 2/3 unused
// for 2D tiles -> zero). This toolchain's builtin takes 6 args (extra int32x8).
__device__ __forceinline__ void tdm_load_wchunk(const unsigned short* gsrc,
                                                unsigned lds_off) {
  unsigned long long ga = (unsigned long long)(uintptr_t)gsrc;  // byte address
  v4u g0;
  g0[0] = 1u;                                    // count=1, user descriptor
  g0[1] = lds_off;                               // lds_addr (bytes)
  g0[2] = (unsigned)(ga & 0xFFFFFFFFu);          // global_addr[31:0]
  g0[3] = (unsigned)((ga >> 32) & 0x1FFFFFFu)    // global_addr[56:32]
          | (2u << 30);                          // type = 2 ("image")
  v8i g1;
  g1[0] = (int)(1u << 16);                       // wg_mask=0, data_size=1 (2B)
  g1[1] = (int)((CC & 0xFFFFu) << 16);           // tensor_dim0 lo16 (=384)
  g1[2] = (int)((CC >> 16) | ((192u & 0xFFFFu) << 16));  // dim0 hi | tensor_dim1 lo (=192)
  g1[3] = (int)((192u >> 16) | (32u << 16));     // dim1 hi | tile_dim0 = 32
  g1[4] = (int)(192u);                           // tile_dim1 = 192, tile_dim2 = 0
  g1[5] = (int)CC;                               // tensor_dim0_stride lo32 (=384)
  g1[6] = 0;                                     // stride0 hi16 | stride1 lo16
  g1[7] = 0;                                     // stride1 hi32
  v4i z4 = {0, 0, 0, 0};
  v8i z8 = {0, 0, 0, 0, 0, 0, 0, 0};
  __builtin_amdgcn_tensor_load_to_lds(g0, g1, z4, z4, z8, 0);
}
#endif

// ---------------------------------------------------------------------------
// Kernel 0: transpose + convert weights to bf16: wt[mat][n][k] = w[k][n]
// ---------------------------------------------------------------------------
__global__ void k_prep(const float* __restrict__ wq, const float* __restrict__ wk,
                       const float* __restrict__ wv, unsigned short* __restrict__ wt) {
  int i = blockIdx.x * blockDim.x + threadIdx.x;
  const int per = DD * CC;                 // 24576
  if (i >= 3 * per) return;
  int mat = i / per;
  int rem = i - mat * per;
  int n = rem / CC;
  int k = rem - n * CC;
  const float* w = (mat == 0) ? wq : (mat == 1) ? wk : wv;
  wt[i] = f2bf(w[k * DD + n]);
}

// ---------------------------------------------------------------------------
// Kernel 1: fused QKV projection + bias + RoPE(angle = t), bf16 outputs.
// 4 waves/block, one 16-row M-tile per wave; the 3 transposed weight matrices
// are streamed through LDS in 12KB K-chunks (double-buffered via the Tensor
// Data Mover when available), shared by all 4 waves.
// Q,K stored (b,t,d) row-major; V stored transposed (b,d,t).
// ---------------------------------------------------------------------------
__global__ void __launch_bounds__(128)
k_qkv(const float* __restrict__ x, const float* __restrict__ bq,
      const float* __restrict__ bk, const float* __restrict__ bv,
      const unsigned short* __restrict__ wt,
      unsigned short* __restrict__ qws, unsigned short* __restrict__ kws,
      unsigned short* __restrict__ vt) {
  // [buf][row = mat*64 + n][k 0..31], row-contiguous 64B
  __shared__ __align__(16) unsigned short wbuf[2][192 * 32];

  const int tid = threadIdx.x;
  const int widx = tid >> 5;
  const int lane = tid & 31;
  const int ln = lane & 15, h = lane >> 4;
  const int wave = (blockIdx.x * blockDim.x + tid) >> 5;   // 0..4095
  const int row0 = wave * 16;               // flat (b*T + t) row
  const int b = row0 >> 9;                  // /512
  const int t0 = row0 & 511;

  constexpr int NCHUNK = CC / 32;           // 12

#if USE_TDM
  if (widx == 0) tdm_load_wchunk(wt, (unsigned)(uintptr_t)&wbuf[0][0]);
#endif

  v8f acc[12] = {};                         // [mat*4 + ntile]

  for (int c = 0; c < NCHUNK; ++c) {
    const int kk = c * 32;

#if USE_TDM
    const unsigned short* wch = &wbuf[c & 1][0];
    if (widx == 0) {
      if (c + 1 < NCHUNK) {
        tdm_load_wchunk(wt + (c + 1) * 32,
                        (unsigned)(uintptr_t)&wbuf[(c + 1) & 1][0]);
        __builtin_amdgcn_s_wait_tensorcnt(1);   // chunk c complete (in-order)
      } else {
        __builtin_amdgcn_s_wait_tensorcnt(0);
      }
    }
    __syncthreads();                            // chunk c visible to all waves
#else
    const unsigned short* wch = &wbuf[0][0];
    __syncthreads();                            // previous chunk consumed
    {
      // cooperative copy: 192 rows x 32 bf16 = 768 uint4, 6 per thread
      uint4* dst = reinterpret_cast<uint4*>(&wbuf[0][0]);
#pragma unroll
      for (int j = 0; j < 6; ++j) {
        int idx = tid + j * 128;                // 0..767
        int r = idx >> 2, q = idx & 3;
        dst[idx] = reinterpret_cast<const uint4*>(wt + r * CC + kk)[q];
      }
    }
    __syncthreads();
#endif

    // ---- A fragment from x (f32 -> bf16 native casts, RNE)
    const float* xrow = x + (size_t)(row0 + ln) * CC + kk;
    const float4* xp = reinterpret_cast<const float4*>(xrow);
    float4 f0 = xp[2 * h], f1 = xp[2 * h + 1];
    float4 f2 = xp[4 + 2 * h], f3 = xp[5 + 2 * h];
    Frag a;
    a.b[0] = (__bf16)f0.x;  a.b[1] = (__bf16)f0.y;
    a.b[2] = (__bf16)f0.z;  a.b[3] = (__bf16)f0.w;
    a.b[4] = (__bf16)f1.x;  a.b[5] = (__bf16)f1.y;
    a.b[6] = (__bf16)f1.z;  a.b[7] = (__bf16)f1.w;
    a.b[8] = (__bf16)f2.x;  a.b[9] = (__bf16)f2.y;
    a.b[10] = (__bf16)f2.z; a.b[11] = (__bf16)f2.w;
    a.b[12] = (__bf16)f3.x; a.b[13] = (__bf16)f3.y;
    a.b[14] = (__bf16)f3.z; a.b[15] = (__bf16)f3.w;

#pragma unroll
    for (int nt = 0; nt < 4; ++nt) {
#pragma unroll
      for (int mat = 0; mat < 3; ++mat) {
        const unsigned short* wrow = wch + (mat * 64 + nt * 16 + ln) * 32;
        Frag bf = load_b_bf16(wrow, h);          // ds_load_b128 x2
        acc[mat * 4 + nt] = wmma_bf16(a.b, bf.b, acc[mat * 4 + nt]);
      }
    }

#if USE_TDM
    __syncthreads();   // all waves done with buf (c&1) before TDM reuses it
#endif
  }

  // RoPE: fres==1 => rotate every (even,odd) pair by angle t.
  float cs[8], sn[8];
#pragma unroll
  for (int r = 0; r < 8; ++r) {
    __sincosf((float)(t0 + r + 8 * h), &sn[r], &cs[r]);
  }

#pragma unroll
  for (int nt = 0; nt < 4; ++nt) {
    const int n = nt * 16 + ln;
    const float bqv = bq[n], bkv = bk[n], bvv = bv[n];
#pragma unroll
    for (int r = 0; r < 8; ++r) {
      const size_t row = (size_t)row0 + r + 8 * h;
      const int trow = t0 + r + 8 * h;
      float qv = acc[0 * 4 + nt][r] + bqv;
      float kv = acc[1 * 4 + nt][r] + bkv;
      float vv = acc[2 * 4 + nt][r] + bvv;
      float qp = __shfl_xor(qv, 1);         // partner element of the pair
      float kp = __shfl_xor(kv, 1);
      float qo, ko;
      if ((n & 1) == 0) {                   // even d: xe*cos - xo*sin
        qo = qv * cs[r] - qp * sn[r];
        ko = kv * cs[r] - kp * sn[r];
      } else {                              // odd d:  xe*sin + xo*cos
        qo = qp * sn[r] + qv * cs[r];
        ko = kp * sn[r] + kv * cs[r];
      }
      qws[row * DD + n] = f2bf(qo);
      kws[row * DD + n] = f2bf(ko);
      vt[((size_t)b * DD + n) * TT + trow] = f2bf(vv);   // V transposed (b,d,t)
    }
  }
}

// ---------------------------------------------------------------------------
// Kernel 2: flash-attention. One wave owns 16 query rows, full d=64,
// streams 32-key blocks with online softmax. P transposed through LDS.
// ---------------------------------------------------------------------------
__global__ void __launch_bounds__(128)
k_attn(const unsigned short* __restrict__ qws, const unsigned short* __restrict__ kws,
       const unsigned short* __restrict__ vt, float* __restrict__ out) {
  __shared__ __align__(16) unsigned short plds[4][16 * 32];

  const int qt = (blockIdx.x * blockDim.x + threadIdx.x) >> 5;   // 0..4095
  const int widx = (threadIdx.x >> 5) & 3;
  const int lane = threadIdx.x & 31;
  const int ln = lane & 15, h = lane >> 4;
  const int b = qt >> 5;                    // T/16 = 32 q-tiles per batch
  const int q0 = (qt & 31) * 16;
  const size_t bT = (size_t)b * TT;

  constexpr float SCALE = 0.05103103630798288f;   // 1/sqrt(384)  (C^-0.5)

  // Q A-fragments (rows q0..q0+15, d split 0..31 / 32..63), resident all loop.
  const unsigned short* qrow = qws + (bT + q0 + ln) * DD;
  Frag qa0 = load_a_bf16(qrow, h);
  Frag qa1 = load_a_bf16(qrow + 32, h);

  float m[8], l[8];
  v8f acc[4] = {};
#pragma unroll
  for (int r = 0; r < 8; ++r) { m[r] = -__builtin_inff(); l[r] = 0.0f; }

  unsigned short* pl = plds[widx];
  const int kbmax = (q0 + 15) >> 5;

  for (int kb = 0; kb <= kbmax; ++kb) {
    const int kc0 = kb << 5;

    // ---- scores S = Q * K^T for 32 keys (two 16x16 tiles)
    v8f s0 = {}, s1 = {};
    {
      const unsigned short* kr0 = kws + (bT + kc0 + ln) * DD;
      Frag kb00 = load_b_bf16(kr0, h);        // d 0..31
      Frag kb01 = load_b_bf16(kr0 + 32, h);   // d 32..63
      s0 = wmma_bf16(qa0.b, kb00.b, s0);
      s0 = wmma_bf16(qa1.b, kb01.b, s0);
      const unsigned short* kr1 = kws + (bT + kc0 + 16 + ln) * DD;
      Frag kb10 = load_b_bf16(kr1, h);
      Frag kb11 = load_b_bf16(kr1 + 32, h);
      s1 = wmma_bf16(qa0.b, kb10.b, s1);
      s1 = wmma_bf16(qa1.b, kb11.b, s1);
    }

    if (kb < kbmax) {  // prefetch next key block (global_prefetch_b8)
      __builtin_prefetch(kws + (bT + kc0 + 32 + ln) * DD, 0, 1);
      __builtin_prefetch(vt + ((size_t)b * DD + ln) * TT + kc0 + 32, 0, 1);
    }

    // ---- online softmax update (row stats live across 16-lane halves)
#pragma unroll
    for (int r = 0; r < 8; ++r) {
      const int tq = q0 + r + 8 * h;
      float v0 = s0[r] * SCALE;
      float v1 = s1[r] * SCALE;
      if (kc0 + ln > tq)      v0 = -__builtin_inff();   // causal mask
      if (kc0 + 16 + ln > tq) v1 = -__builtin_inff();
      float mx = fmaxf(v0, v1);
#pragma unroll
      for (int d = 1; d < 16; d <<= 1) mx = fmaxf(mx, __shfl_xor(mx, d));
      const float mn = fmaxf(m[r], mx);
      const float corr = __expf(m[r] - mn);             // -inf -> 0 first block
      const float p0 = __expf(v0 - mn);
      const float p1 = __expf(v1 - mn);
      float sum = p0 + p1;
#pragma unroll
      for (int d = 1; d < 16; d <<= 1) sum += __shfl_xor(sum, d);
      l[r] = l[r] * corr + sum;
      m[r] = mn;
#pragma unroll
      for (int nt = 0; nt < 4; ++nt) acc[nt][r] *= corr;
      // stage P (bf16) in LDS as [t_q][t_k] for the A-layout reload
      pl[(r + 8 * h) * 32 + ln] = f2bf(p0);
      pl[(r + 8 * h) * 32 + 16 + ln] = f2bf(p1);
    }

    // wave-local LDS RAW: DS ops are in-order per wave; just drain dscnt.
    asm volatile("s_wait_dscnt 0" ::: "memory");

    Frag pa = load_a_bf16(pl + ln * 32, h);   // P as 16x32 A fragment

    // ---- acc += P * V  (V transposed => contiguous B fragments)
#pragma unroll
    for (int nt = 0; nt < 4; ++nt) {
      const unsigned short* vrow = vt + ((size_t)b * DD + nt * 16 + ln) * TT + kc0;
      Frag vb = load_b_bf16(vrow, h);
      acc[nt] = wmma_bf16(pa.b, vb.b, acc[nt]);
    }
  }

  // ---- normalize and store f32 output (b, t, d)
#pragma unroll
  for (int nt = 0; nt < 4; ++nt) {
    const int n = nt * 16 + ln;
#pragma unroll
    for (int r = 0; r < 8; ++r) {
      const size_t row = bT + q0 + r + 8 * h;
      out[row * DD + n] = acc[nt][r] / l[r];
    }
  }
}

// ---------------------------------------------------------------------------
extern "C" void kernel_launch(void* const* d_in, const int* in_sizes, int n_in,
                              void* d_out, int out_size, void* d_ws, size_t ws_size,
                              hipStream_t stream) {
  const float* x  = (const float*)d_in[0];
  const float* wq = (const float*)d_in[1];
  const float* bq = (const float*)d_in[2];
  const float* wk = (const float*)d_in[3];
  const float* bk = (const float*)d_in[4];
  const float* wv = (const float*)d_in[5];
  const float* bv = (const float*)d_in[6];

  // Workspace layout (all bf16 stored as u16):
  //   wt : 3 * 64*384   (n,k)        transposed weights
  //   q  : 128*512*64   (b,t,d)      post-RoPE Q
  //   k  : 128*512*64   (b,t,d)      post-RoPE K
  //   vt : 128*64*512   (b,d,t)      V transposed
  unsigned short* wt  = (unsigned short*)d_ws;
  unsigned short* qws = wt + 3 * DD * CC;
  unsigned short* kws = qws + (size_t)NB * TT * DD;
  unsigned short* vtw = kws + (size_t)NB * TT * DD;

  k_prep<<<(3 * DD * CC + 255) / 256, 256, 0, stream>>>(wq, wk, wv, wt);

  // 4096 row-tiles, 4 waves (128 threads) per block
  k_qkv<<<1024, 128, 0, stream>>>(x, bq, bk, bv, wt, qws, kws, vtw);

  // 4096 q-tiles, 4 waves per block
  k_attn<<<1024, 128, 0, stream>>>(qws, kws, vtw, (float*)d_out);
}
